// HybridNNP_69861938036852
// MI455X (gfx1250) — matrix operand order repaired and checked by reading, not compiled, not run on hardware
//
#include <hip/hip_runtime.h>
#include <hip/hip_bf16.h>

// ---------------------------------------------------------------------------
// HybridNNP for MI455X (gfx1250, wave32, WMMA).
// Pipeline: init -> species bucketing -> weight transpose/pad to bf16 ->
//           AEV (radial+angular, LDS atomics) -> per-species MLP via
//           v_wmma_f32_16x16x32_bf16 -> scalar energy.
// ---------------------------------------------------------------------------

typedef __attribute__((ext_vector_type(16))) __bf16 v16bf;
typedef __attribute__((ext_vector_type(8)))  __bf16 v8bf;
typedef __attribute__((ext_vector_type(8)))  float  v8f;

#define NML   3072
#define KRL   96
#define KAL   32
#define KPAD  1024      // 1008 padded
#define MAXSLOT 3200    // 3072 + 7*15 pad, rounded
#define MAXTILE 200

// species map: atomic number -> species index {1,6,7,8,16,9,17} -> {0..6}
__constant__ int c_spmap[18] = {0,0,0,0,0,0,1,2,3,5,0,0,0,0,0,0,4,6};
__constant__ float c_shfA[4] = {0.9f, 1.55f, 2.2f, 2.85f};

// padded transposed-weight element counts per layer (all species)
#define E0 (7L*256*1024)
#define E1 (7L*192*256)
#define E2 (7L*160*192)
#define E3 (7L*16*160)

// ---- workspace byte offsets (all 256-aligned where vector loads occur) ----
#define OFF_POS    0u          // 3072*3*4
#define OFF_SPEC   36864u      // 3072*4
#define OFF_CNT    49152u      // 7*4
#define OFF_CUR    49184u      // 7*4
#define OFF_BASE   49216u      // 7*4
#define OFF_ACC    49248u      // 4
#define OFF_ORDER  49280u      // 3200*4
#define OFF_TSPEC  62080u      // 200*4
#define OFF_AEV    65536u      // 3072*1024*2 = 6291456
#define OFF_W0     6356992u    // E0*2 = 3670016
#define OFF_W1     10027008u   // E1*2 = 688128
#define OFF_W2     10715136u   // E2*2 = 430080
#define OFF_W3     11145216u   // E3*2 = 35840

__device__ inline float celu01(float x) {
    return x > 0.f ? x : 0.1f * (__expf(10.f * x) - 1.f);
}

__device__ inline v16bf pack16(v8bf lo, v8bf hi) {
    v16bf r;
#pragma unroll
    for (int t = 0; t < 8; ++t) { r[t] = lo[t]; r[t + 8] = hi[t]; }
    return r;
}
// A fragment (16-bit, 16x32): lane<16 holds M=lane, K = kc+{0..15};
// lane>=16 holds M=lane-16, K = kc+{16..31}.  p already includes khalfA.
__device__ inline v16bf ldA(const __bf16* p) {
    v8bf lo = *(const v8bf*)p;
    v8bf hi = *(const v8bf*)(p + 8);
    return pack16(lo, hi);
}
// B fragment (16-bit, 32x16): lane holds N=lane&15, K = kc + (lane<16?0:16) + 0..15
__device__ inline v16bf ldB(const __bf16* p) {
    v8bf lo = *(const v8bf*)p;
    v8bf hi = *(const v8bf*)(p + 8);
    return pack16(lo, hi);
}

// ---------------------------------------------------------------------------
__global__ void k_init(const float* __restrict__ positions,
                       const int* __restrict__ indices,
                       const int* __restrict__ atnum,
                       float* pos, int* spec, int* cnt, int* cur, int* base,
                       float* acc, int* order, int* tspec) {
    int t = blockIdx.x * blockDim.x + threadIdx.x;
    if (t < 7) { cnt[t] = 0; cur[t] = 0; base[t] = 0; }
    if (t == 0) *acc = 0.f;
    if (t < MAXSLOT) order[t] = -1;
    if (t < MAXTILE) tspec[t] = -1;
    if (t < NML) {
        int src = indices[t];
        pos[t * 3 + 0] = positions[src * 3 + 0] * 10.f;
        pos[t * 3 + 1] = positions[src * 3 + 1] * 10.f;
        pos[t * 3 + 2] = positions[src * 3 + 2] * 10.f;
        spec[t] = c_spmap[atnum[t] & 31];
    }
}

__global__ void k_count(const int* __restrict__ spec, int* cnt) {
    int i = blockIdx.x * blockDim.x + threadIdx.x;
    if (i < NML) atomicAdd(&cnt[spec[i]], 1);
}

__global__ void k_offsets(const int* __restrict__ cnt, int* base, int* tspec) {
    if (threadIdx.x == 0 && blockIdx.x == 0) {
        int off = 0;
        for (int s = 0; s < 7; ++s) {
            base[s] = off;
            int ct = (cnt[s] + 15) >> 4;
            for (int t = 0; t < ct; ++t) tspec[(off >> 4) + t] = s;
            off += ct << 4;
        }
    }
}

__global__ void k_scatter(const int* __restrict__ spec, const int* __restrict__ base,
                          int* cur, int* order) {
    int i = blockIdx.x * blockDim.x + threadIdx.x;
    if (i < NML) {
        int s = spec[i];
        int p = atomicAdd(&cur[s], 1);
        order[base[s] + p] = i;
    }
}

// transpose + zero-pad weights to bf16, layout [s][n][kpad] (K contiguous per row)
__global__ void k_convw(const float* __restrict__ W0, const float* __restrict__ W1,
                        const float* __restrict__ W2, const float* __restrict__ W3,
                        __bf16* Wt0, __bf16* Wt1, __bf16* Wt2, __bf16* Wt3) {
    long t = (long)blockIdx.x * blockDim.x + threadIdx.x;
    if (t < E0) {
        long s = t / (256 * 1024), r = t % (256 * 1024), n = r / 1024, k = r % 1024;
        float v = (k < 1008) ? W0[(s * 1008 + k) * 256 + n] : 0.f;
        Wt0[t] = (__bf16)v;
    } else if (t < E0 + E1) {
        long u = t - E0, s = u / (192 * 256), r = u % (192 * 256), n = r / 256, k = r % 256;
        Wt1[u] = (__bf16)W1[(s * 256 + k) * 192 + n];
    } else if (t < E0 + E1 + E2) {
        long u = t - E0 - E1, s = u / (160 * 192), r = u % (160 * 192), n = r / 192, k = r % 192;
        Wt2[u] = (__bf16)W2[(s * 192 + k) * 160 + n];
    } else if (t < E0 + E1 + E2 + E3) {
        long u = t - E0 - E1 - E2, s = u / (16 * 160), r = u % (16 * 160), n = r / 160, k = r % 160;
        float v = (n == 0) ? W3[s * 160 + k] : 0.f;
        Wt3[u] = (__bf16)v;
    }
}

// ---------------------------------------------------------------------------
// AEV: one workgroup (128 thr) per atom. Cutoff-based capped neighbor list
// (density << caps of 96/32, so this matches the reference's top-k selection).
__global__ __launch_bounds__(128) void k_aev(const float* __restrict__ pos,
                                             const int* __restrict__ spec,
                                             __bf16* __restrict__ aev) {
    const int i = blockIdx.x, tid = threadIdx.x, nthr = blockDim.x;
    __shared__ float s_dR[KRL]; __shared__ int s_jR[KRL];
    __shared__ float s_dA[KAL]; __shared__ float s_vA[KAL][3]; __shared__ int s_sA[KAL];
    __shared__ float s_rad[112]; __shared__ float s_ang[896];
    __shared__ int s_cntR, s_cntA;
    if (tid == 0) { s_cntR = 0; s_cntA = 0; }
    for (int f = tid; f < 112; f += nthr) s_rad[f] = 0.f;
    for (int f = tid; f < 896; f += nthr) s_ang[f] = 0.f;
    const float px = pos[i * 3], py = pos[i * 3 + 1], pz = pos[i * 3 + 2];
    __syncthreads();

    for (int j = tid; j < NML; j += nthr) {
        if (j == i) continue;
        float dx = pos[j * 3] - px, dy = pos[j * 3 + 1] - py, dz = pos[j * 3 + 2] - pz;
        float d = sqrtf(dx * dx + dy * dy + dz * dz + 1e-12f);
        if (d < 5.1f) {
            int sl = atomicAdd(&s_cntR, 1);
            if (sl < KRL) { s_dR[sl] = d; s_jR[sl] = j; }
        }
    }
    __syncthreads();
    int nR = min(s_cntR, KRL);
    for (int n = tid; n < nR; n += nthr) {
        float d = s_dR[n];
        if (d < 3.5f) {
            int sl = atomicAdd(&s_cntA, 1);
            if (sl < KAL) {
                int j = s_jR[n];
                s_dA[sl] = d;
                s_vA[sl][0] = pos[j * 3] - px;
                s_vA[sl][1] = pos[j * 3 + 1] - py;
                s_vA[sl][2] = pos[j * 3 + 2] - pz;
                s_sA[sl] = spec[j];
            }
        }
    }
    __syncthreads();
    // radial: 0.25*exp(-19.7*(d-ShfR)^2)*fc  into rad[s_j][q]
    for (int w = tid; w < nR * 16; w += nthr) {
        int n = w >> 4, q = w & 15;
        float d = s_dR[n];
        float fc = 0.5f * __cosf(3.14159265358979f * d / 5.1f) + 0.5f;
        float tt = d - (0.8f + 0.26875f * q);
        atomicAdd(&s_rad[spec[s_jR[n]] * 16 + q], 0.25f * __expf(-19.7f * tt * tt) * fc);
    }
    int nA = min(s_cntA, KAL);
    int npair = nA * (nA - 1) / 2;
    __syncthreads();
    for (int p = tid; p < npair; p += nthr) {
        int a = 0, rem = p;
        while (rem >= nA - 1 - a) { rem -= nA - 1 - a; a++; }
        int b = a + 1 + rem;
        float d1 = s_dA[a], d2 = s_dA[b];
        float dot = s_vA[a][0] * s_vA[b][0] + s_vA[a][1] * s_vA[b][1] + s_vA[a][2] * s_vA[b][2];
        float cth = 0.95f * dot / (d1 * d2);
        cth = fminf(0.999999f, fmaxf(-0.999999f, cth));
        float theta = acosf(cth);
        float fc1 = 0.5f * __cosf(3.14159265358979f * d1 / 3.5f) + 0.5f;
        float fc2 = 0.5f * __cosf(3.14159265358979f * d2 / 3.5f) + 0.5f;
        float pref = 2.f * fc1 * fc2;
        float ravg = 0.5f * (d1 + d2);
        int sa = s_sA[a], sb = s_sA[b];
        int lo = min(sa, sb), hi = max(sa, sb);
        int pidx = lo * 7 - (lo * (lo - 1)) / 2 + (hi - lo);
        float f1[8];
#pragma unroll
        for (int zi = 0; zi < 8; ++zi) {
            float c = 0.5f * (1.f + __cosf(theta - (zi + 0.5f) * 0.39269908169872f));
            f1[zi] = __powf(c, 14.1f);
        }
#pragma unroll
        for (int ai = 0; ai < 4; ++ai) {
            float rr = ravg - c_shfA[ai];
            float f2 = pref * __expf(-12.5f * rr * rr);
#pragma unroll
            for (int zi = 0; zi < 8; ++zi)
                atomicAdd(&s_ang[pidx * 32 + ai * 8 + zi], f2 * f1[zi]);
        }
    }
    __syncthreads();
    __bf16* out = aev + (size_t)i * KPAD;
    for (int f = tid; f < KPAD; f += nthr) {
        float v = 0.f;
        if (f < 112) v = s_rad[f];
        else if (f < 1008) v = s_ang[f - 112];
        out[f] = (__bf16)v;
    }
}

// ---------------------------------------------------------------------------
// MLP: one block = one 16-atom same-species tile; 8 waves split N-tiles.
__global__ __launch_bounds__(256) void k_mlp(const __bf16* __restrict__ aev,
                                             const int* __restrict__ order,
                                             const int* __restrict__ tspec,
                                             const __bf16* __restrict__ W0t,
                                             const __bf16* __restrict__ W1t,
                                             const __bf16* __restrict__ W2t,
                                             const __bf16* __restrict__ W3t,
                                             const float* __restrict__ b0,
                                             const float* __restrict__ b1,
                                             const float* __restrict__ b2,
                                             const float* __restrict__ b3,
                                             const float* __restrict__ self_e,
                                             float* acc) {
    const int tile = blockIdx.x;
    const int s = tspec[tile];
    if (s < 0) return;

    __shared__ __bf16 Xs[16 * 1024];
    __shared__ __bf16 H1[16 * 256];
    __shared__ __bf16 H2[16 * 192];
    __shared__ __bf16 H3[16 * 160];
    __shared__ int atom16[16];

    const int tid = threadIdx.x;
    if (tid < 16) atom16[tid] = order[tile * 16 + tid];
    __syncthreads();

    // stage the 16 AEV rows into LDS (zeros for padding rows)
    for (int c = tid; c < 16 * 128; c += 256) {
        int m = c >> 7, ch = c & 127;
        v8bf v = {};
        int idx = atom16[m];
        if (idx >= 0) v = *(const v8bf*)(aev + (size_t)idx * KPAD + ch * 8);
        *(v8bf*)(Xs + m * 1024 + ch * 8) = v;
    }
    __syncthreads();

    const int wave = tid >> 5, lane = tid & 31;
    const int col = lane & 15;                  // = A row M = B col N = C/D col
    const int khA = (lane & 16) ? 16 : 0;
    const int khB = (lane & 16) ? 16 : 0;
    const int rbase = (lane & 16) ? 8 : 0;      // C/D rows r+rbase

    // ---- layer 0: 16x1024 @ 1024x256 ----
    {
        const __bf16* w = W0t + (size_t)s * 256 * 1024;
        const __bf16* arow = Xs + col * 1024 + khA;
        for (int nt = wave; nt < 16; nt += 8) {
            v8f d = {};
            const __bf16* brow = w + (size_t)(nt * 16 + col) * 1024 + khB;
            for (int kc = 0; kc < 1024; kc += 32) {
                __builtin_prefetch(brow + kc + 128, 0, 1);
                v16bf a = ldA(arow + kc);
                v16bf b = ldB(brow + kc);
                d = __builtin_amdgcn_wmma_f32_16x16x32_bf16(false, a, false, b,
                                                            (short)0, d, false, false);
            }
            float bias = b0[s * 256 + nt * 16 + col];
#pragma unroll
            for (int r = 0; r < 8; ++r)
                H1[(r + rbase) * 256 + nt * 16 + col] = (__bf16)celu01(d[r] + bias);
        }
    }
    __syncthreads();

    // ---- layer 1: 16x256 @ 256x192 ----
    {
        const __bf16* w = W1t + (size_t)s * 192 * 256;
        const __bf16* arow = H1 + col * 256 + khA;
        for (int nt = wave; nt < 12; nt += 8) {
            v8f d = {};
            const __bf16* brow = w + (size_t)(nt * 16 + col) * 256 + khB;
            for (int kc = 0; kc < 256; kc += 32) {
                v16bf a = ldA(arow + kc);
                v16bf b = ldB(brow + kc);
                d = __builtin_amdgcn_wmma_f32_16x16x32_bf16(false, a, false, b,
                                                            (short)0, d, false, false);
            }
            float bias = b1[s * 192 + nt * 16 + col];
#pragma unroll
            for (int r = 0; r < 8; ++r)
                H2[(r + rbase) * 192 + nt * 16 + col] = (__bf16)celu01(d[r] + bias);
        }
    }
    __syncthreads();

    // ---- layer 2: 16x192 @ 192x160 ----
    {
        const __bf16* w = W2t + (size_t)s * 160 * 192;
        const __bf16* arow = H2 + col * 192 + khA;
        for (int nt = wave; nt < 10; nt += 8) {
            v8f d = {};
            const __bf16* brow = w + (size_t)(nt * 16 + col) * 192 + khB;
            for (int kc = 0; kc < 192; kc += 32) {
                v16bf a = ldA(arow + kc);
                v16bf b = ldB(brow + kc);
                d = __builtin_amdgcn_wmma_f32_16x16x32_bf16(false, a, false, b,
                                                            (short)0, d, false, false);
            }
            float bias = b2[s * 160 + nt * 16 + col];
#pragma unroll
            for (int r = 0; r < 8; ++r)
                H3[(r + rbase) * 160 + nt * 16 + col] = (__bf16)celu01(d[r] + bias);
        }
    }
    __syncthreads();

    // ---- layer 3: 16x160 @ 160x16 (only col 0 valid) ----
    if (wave == 0) {
        const __bf16* w = W3t + (size_t)s * 16 * 160;
        const __bf16* arow = H3 + col * 160 + khA;
        const __bf16* brow = w + (size_t)col * 160 + khB;
        v8f d = {};
        for (int kc = 0; kc < 160; kc += 32) {
            v16bf a = ldA(arow + kc);
            v16bf b = ldB(brow + kc);
            d = __builtin_amdgcn_wmma_f32_16x16x32_bf16(false, a, false, b,
                                                        (short)0, d, false, false);
        }
        if (col == 0) {
            float bb = b3[s] + self_e[s];
            float part = 0.f;
#pragma unroll
            for (int r = 0; r < 8; ++r) {
                int m = r + rbase;
                if (atom16[m] >= 0) part += d[r] + bb;
            }
            atomicAdd(acc, part);
        }
    }
}

__global__ void k_final(const float* __restrict__ acc, float* __restrict__ out) {
    if (threadIdx.x == 0 && blockIdx.x == 0) out[0] = acc[0] * 2625.5f;
}

// ---------------------------------------------------------------------------
extern "C" void kernel_launch(void* const* d_in, const int* in_sizes, int n_in,
                              void* d_out, int out_size, void* d_ws, size_t ws_size,
                              hipStream_t stream) {
    (void)in_sizes; (void)n_in; (void)out_size; (void)ws_size;
    const float* positions = (const float*)d_in[0];
    const int*   indices   = (const int*)d_in[1];
    const int*   atnum     = (const int*)d_in[2];
    const float* W0 = (const float*)d_in[3];
    const float* b0 = (const float*)d_in[4];
    const float* W1 = (const float*)d_in[5];
    const float* b1 = (const float*)d_in[6];
    const float* W2 = (const float*)d_in[7];
    const float* b2 = (const float*)d_in[8];
    const float* W3 = (const float*)d_in[9];
    const float* b3 = (const float*)d_in[10];
    const float* self_e = (const float*)d_in[11];
    float* out = (float*)d_out;

    char* ws = (char*)d_ws;
    float*  pos   = (float*)(ws + OFF_POS);
    int*    spec  = (int*)(ws + OFF_SPEC);
    int*    cnt   = (int*)(ws + OFF_CNT);
    int*    cur   = (int*)(ws + OFF_CUR);
    int*    base  = (int*)(ws + OFF_BASE);
    float*  acc   = (float*)(ws + OFF_ACC);
    int*    order = (int*)(ws + OFF_ORDER);
    int*    tsp   = (int*)(ws + OFF_TSPEC);
    __bf16* aev   = (__bf16*)(ws + OFF_AEV);
    __bf16* Wt0   = (__bf16*)(ws + OFF_W0);
    __bf16* Wt1   = (__bf16*)(ws + OFF_W1);
    __bf16* Wt2   = (__bf16*)(ws + OFF_W2);
    __bf16* Wt3   = (__bf16*)(ws + OFF_W3);

    k_init<<<(MAXSLOT + 255) / 256, 256, 0, stream>>>(positions, indices, atnum, pos, spec,
                                                      cnt, cur, base, acc, order, tsp);
    k_count<<<(NML + 255) / 256, 256, 0, stream>>>(spec, cnt);
    k_offsets<<<1, 32, 0, stream>>>(cnt, base, tsp);
    k_scatter<<<(NML + 255) / 256, 256, 0, stream>>>(spec, base, cur, order);
    long eall = E0 + E1 + E2 + E3;
    k_convw<<<(int)((eall + 255) / 256), 256, 0, stream>>>(W0, W1, W2, W3, Wt0, Wt1, Wt2, Wt3);
    k_aev<<<NML, 128, 0, stream>>>(pos, spec, aev);
    k_mlp<<<MAXTILE, 256, 0, stream>>>(aev, order, tsp, Wt0, Wt1, Wt2, Wt3,
                                       b0, b1, b2, b3, self_e, acc);
    k_final<<<1, 32, 0, stream>>>(acc, out);
}
